// LiquidCell_84078279787109
// MI455X (gfx1250) — matrix operand order repaired and checked by reading, not compiled, not run on hardware
//
#include <hip/hip_runtime.h>
#include <hip/hip_bf16.h>

// LiquidCell: pre-converted bf16 WMMA projections + chunked associative scan.
// B=8, T=4096, IN=1024, HIDDEN=1024.

#define B_DIM   8
#define T_DIM   4096
#define K_DIM   1024
#define H_DIM   1024
#define M_DIM   (B_DIM * T_DIM)   // 32768 rows
#define NCHUNK  32
#define CLEN    (T_DIM / NCHUNK)  // 128

typedef __attribute__((ext_vector_type(16))) __bf16 v16bf;
typedef __attribute__((ext_vector_type(8)))  float  v8f;

union BFragU { v16bf v; uint4 u[2]; };

__device__ __forceinline__ unsigned int f2bf_bits(float f) {
  union { float f; unsigned int u; } c; c.f = f;
  unsigned int u = c.u;
  u += 0x7fffu + ((u >> 16) & 1u);   // round-to-nearest-even
  return u >> 16;
}
__device__ __forceinline__ unsigned int pack2bf(float lo, float hi) {
  return f2bf_bits(lo) | (f2bf_bits(hi) << 16);
}

// Branchless numerically-stable softplus using fast transcendentals.
__device__ __forceinline__ float softplus_fast(float x) {
  return fmaxf(x, 0.f) + __logf(1.f + __expf(-fabsf(x)));
}

// Fast tanh: clamp so exp stays finite, then (e-1)/(e+1).
__device__ __forceinline__ float tanh_fast(float x) {
  const float xc = fminf(fmaxf(x, -9.f), 9.f);
  const float e = __expf(2.f * xc);
  return __fdividef(e - 1.f, e + 1.f);
}

// ---------------------------------------------------------------------------
// f32 -> bf16 bulk converter: each thread converts 8 elements (one uint4 out).
// ---------------------------------------------------------------------------
__global__ __launch_bounds__(256)
void f32_to_bf16_kernel(const float* __restrict__ in, unsigned int* __restrict__ out) {
  const size_t i = (size_t)blockIdx.x * blockDim.x + threadIdx.x;  // 8 elems per thread
  const float4 f0 = ((const float4*)in)[2 * i];
  const float4 f1 = ((const float4*)in)[2 * i + 1];
  uint4 o;
  o.x = pack2bf(f0.x, f0.y);
  o.y = pack2bf(f0.z, f0.w);
  o.z = pack2bf(f1.x, f1.y);
  o.w = pack2bf(f1.z, f1.w);
  ((uint4*)out)[i] = o;
}

// A-fragment (16x32 bf16): lane L: M = L%16; elems 0..7 = K[k0..k0+8),
// elems 8..15 = K[k0+16..k0+24) where k0 = kt + (L/16)*8.
__device__ __forceinline__ v16bf load_a_frag(const __bf16* __restrict__ row, int k0) {
  BFragU fr;
  fr.u[0] = *(const uint4*)(row + k0);
  fr.u[1] = *(const uint4*)(row + k0 + 16);
  return fr.v;
}

// B-fragment (32x16 bf16): lane L: N = L%16; elems 0..7 = K[k0..k0+8),
// elems 8..15 = K[k0+16..k0+24) where k0 = kt + (L/16)*8.
__device__ __forceinline__ v16bf load_b_frag(const __bf16* __restrict__ row, int k0) {
  BFragU fr;
  fr.u[0] = *(const uint4*)(row + k0);
  fr.u[1] = *(const uint4*)(row + k0 + 16);
  return fr.v;
}

#define WMMA_BF16(A, Bm, C) \
  __builtin_amdgcn_wmma_f32_16x16x32_bf16(false, (A), false, (Bm), (short)0, (C), false, false)

// ---------------------------------------------------------------------------
// Kernel: dual GEMM + fused epilogue, all-bf16 operands from ws.
// Wave tile = 64M x 32N for BOTH weight matrices: 16 WMMAs per K-step,
// 16 b128 loads per K-step (1.0 vmem/wmma), A & B frags each reused 4x.
// A-fragments are software-pipelined one K-step ahead (2-stage, two buffers)
// so no WMMA group waits on a just-issued load.
// Workgroup: 8 waves = 2 (M) x 4 (N) -> 128M x 128N. grid = (H/128, M/128).
// Writes At -> ws, bt -> d_out (reused as scratch; overwritten by tanh later).
// ---------------------------------------------------------------------------
__global__ __launch_bounds__(256)
void liquid_proj_wmma(const __bf16* __restrict__ xb,
                      const __bf16* __restrict__ Wdb,
                      const __bf16* __restrict__ Wbb,
                      const float* __restrict__ bd,
                      const float* __restrict__ bb,
                      const float* __restrict__ A_log,
                      float* __restrict__ Atp, float* __restrict__ btp) {
  const int wave = threadIdx.x >> 5;
  const int lane = threadIdx.x & 31;
  const int half = lane >> 4;
  const int l16  = lane & 15;

  const int Mblk = blockIdx.y * 128 + (wave & 1) * 64;   // 2 waves along M
  const int Nblk = blockIdx.x * 128 + (wave >> 1) * 32;  // 4 waves along N

  const v8f vzero = {0.f, 0.f, 0.f, 0.f, 0.f, 0.f, 0.f, 0.f};
  v8f accD[4][2], accB[4][2];
#pragma unroll
  for (int mi = 0; mi < 4; ++mi)
#pragma unroll
    for (int ni = 0; ni < 2; ++ni) { accD[mi][ni] = vzero; accB[mi][ni] = vzero; }

  const __bf16* xr[4];
#pragma unroll
  for (int mi = 0; mi < 4; ++mi)
    xr[mi] = xb + (size_t)(Mblk + mi * 16 + l16) * K_DIM;

  const __bf16* wdr[2] = { Wdb + (size_t)(Nblk      + l16) * K_DIM,
                           Wdb + (size_t)(Nblk + 16 + l16) * K_DIM };
  const __bf16* wbr[2] = { Wbb + (size_t)(Nblk      + l16) * K_DIM,
                           Wbb + (size_t)(Nblk + 16 + l16) * K_DIM };

  const int kAoff = half * 8;    // A: two 8-elem chunks at kA, kA+16
  const int kBoff = half * 8;    // B: two 8-elem chunks at kB, kB+16

  // Prologue: A-frags for K-step 0.
  v16bf aA[4], aB[4];
#pragma unroll
  for (int mi = 0; mi < 4; ++mi) aA[mi] = load_a_frag(xr[mi], kAoff);

  // Two K-steps (64 K) per iteration; A prefetched one step ahead.
  // The very last prefetch (kt=1024) reads into the adjacent Wdb region of
  // the same ws allocation and is never consumed — safe.
  for (int kt = 0; kt < K_DIM; kt += 64) {
    {
      const int kB = kt + kBoff;
      v16bf d0 = load_b_frag(wdr[0], kB);
      v16bf d1 = load_b_frag(wdr[1], kB);
      v16bf e0 = load_b_frag(wbr[0], kB);
      v16bf e1 = load_b_frag(wbr[1], kB);
      const int kAn = kt + 32 + kAoff;
#pragma unroll
      for (int mi = 0; mi < 4; ++mi) aB[mi] = load_a_frag(xr[mi], kAn);
#pragma unroll
      for (int mi = 0; mi < 4; ++mi) {
        accD[mi][0] = WMMA_BF16(aA[mi], d0, accD[mi][0]);
        accD[mi][1] = WMMA_BF16(aA[mi], d1, accD[mi][1]);
        accB[mi][0] = WMMA_BF16(aA[mi], e0, accB[mi][0]);
        accB[mi][1] = WMMA_BF16(aA[mi], e1, accB[mi][1]);
      }
    }
    {
      const int kB = kt + 32 + kBoff;
      v16bf d0 = load_b_frag(wdr[0], kB);
      v16bf d1 = load_b_frag(wdr[1], kB);
      v16bf e0 = load_b_frag(wbr[0], kB);
      v16bf e1 = load_b_frag(wbr[1], kB);
      const int kAn = kt + 64 + kAoff;
#pragma unroll
      for (int mi = 0; mi < 4; ++mi) aA[mi] = load_a_frag(xr[mi], kAn);
#pragma unroll
      for (int mi = 0; mi < 4; ++mi) {
        accD[mi][0] = WMMA_BF16(aB[mi], d0, accD[mi][0]);
        accD[mi][1] = WMMA_BF16(aB[mi], d1, accD[mi][1]);
        accB[mi][0] = WMMA_BF16(aB[mi], e0, accB[mi][0]);
        accB[mi][1] = WMMA_BF16(aB[mi], e1, accB[mi][1]);
      }
    }
  }

  // Fused epilogue (branchless):
  //   delta = softplus(d + bd); bt = delta*(p + bb); At = exp(-delta*exp(A_log))
#pragma unroll
  for (int ni = 0; ni < 2; ++ni) {
    const int n = Nblk + ni * 16 + l16;
    const float bdv  = bd[n];
    const float bbv  = bb[n];
    const float Adec = __expf(A_log[n]);
#pragma unroll
    for (int mi = 0; mi < 4; ++mi) {
#pragma unroll
      for (int r = 0; r < 8; ++r) {
        const int m = Mblk + mi * 16 + half * 8 + r;   // C/D: lanes16-31 -> M+8
        const float draw = accD[mi][ni][r] + bdv;
        const float praw = accB[mi][ni][r] + bbv;
        const float delta = softplus_fast(draw);
        const float at = __expf(-delta * Adec);
        const size_t o = (size_t)m * H_DIM + n;
        Atp[o] = at;
        btp[o] = delta * praw;
      }
    }
  }
}

// ---------------------------------------------------------------------------
// Kernel: per-(b,h,chunk) affine composition over CLEN steps.
// ---------------------------------------------------------------------------
__global__ __launch_bounds__(256)
void chunk_reduce(const float* __restrict__ At, const float* __restrict__ bt,
                  float* __restrict__ ca, float* __restrict__ cb) {
  const int tid = blockIdx.x * blockDim.x + threadIdx.x;  // [0, B*H*NCHUNK)
  const int h = tid & (H_DIM - 1);
  const int c = (tid >> 10) & (NCHUNK - 1);
  const int b = tid >> 15;
  size_t base = ((size_t)(b * T_DIM + c * CLEN)) * H_DIM + h;
  float a = 1.f, bbv = 0.f;
#pragma unroll 4
  for (int i = 0; i < CLEN; ++i) {
    const float Av = At[base];
    const float Bv = bt[base];
    bbv = fmaf(Av, bbv, Bv);   // combine((a,bbv),(Av,Bv))
    a *= Av;
    base += H_DIM;
  }
  ca[tid] = a;
  cb[tid] = bbv;
}

// ---------------------------------------------------------------------------
// Kernel: scan over the 32 chunks per (b,h); carry-in h for each chunk.
// ---------------------------------------------------------------------------
__global__ __launch_bounds__(256)
void chunk_scan(const float* __restrict__ ca, const float* __restrict__ cb,
                const float* __restrict__ h0, float* __restrict__ carry) {
  const int tid = blockIdx.x * blockDim.x + threadIdx.x;  // [0, B*H)
  const int h = tid & (H_DIM - 1);
  const int b = tid >> 10;
  float cur = h0[b * H_DIM + h];
#pragma unroll
  for (int c = 0; c < NCHUNK; ++c) {
    const int idx = ((b * NCHUNK + c) << 10) + h;
    carry[idx] = cur;
    cur = fmaf(ca[idx], cur, cb[idx]);
  }
}

// ---------------------------------------------------------------------------
// Kernel: re-run the recurrence within each chunk from carry-in, write tanh.
// ---------------------------------------------------------------------------
__global__ __launch_bounds__(256)
void chunk_apply(const float* __restrict__ At, const float* __restrict__ carry,
                 float* __restrict__ out /* holds bt on entry */) {
  const int tid = blockIdx.x * blockDim.x + threadIdx.x;  // [0, B*H*NCHUNK)
  const int h = tid & (H_DIM - 1);
  const int c = (tid >> 10) & (NCHUNK - 1);
  const int b = tid >> 15;
  size_t base = ((size_t)(b * T_DIM + c * CLEN)) * H_DIM + h;
  float hv = carry[tid];
#pragma unroll 4
  for (int i = 0; i < CLEN; ++i) {
    hv = fmaf(At[base], hv, out[base]);  // out[base] currently holds bt
    out[base] = tanh_fast(hv);
    base += H_DIM;
  }
}

extern "C" void kernel_launch(void* const* d_in, const int* in_sizes, int n_in,
                              void* d_out, int out_size, void* d_ws, size_t ws_size,
                              hipStream_t stream) {
  const float* x     = (const float*)d_in[0];
  const float* h0    = (const float*)d_in[1];
  const float* Wd    = (const float*)d_in[2];
  const float* bd    = (const float*)d_in[3];
  const float* Wb    = (const float*)d_in[4];
  const float* bb    = (const float*)d_in[5];
  const float* A_log = (const float*)d_in[6];
  float* out = (float*)d_out;

  // Workspace layout:
  //   At   : M*H f32                (134,217,728 B)
  //   xb   : M*K bf16               ( 67,108,864 B)
  //   Wdb  : H*K bf16               (  2,097,152 B)
  //   Wbb  : H*K bf16               (  2,097,152 B)
  //   ca, cb, carry : B*H*NCHUNK f32 (3 x 1,048,576 B)
  char* ws = (char*)d_ws;
  float*  At  = (float*)ws;
  __bf16* xbp = (__bf16*)(ws + (size_t)M_DIM * H_DIM * sizeof(float));
  __bf16* Wdb = xbp + (size_t)M_DIM * K_DIM;
  __bf16* Wbb = Wdb + (size_t)H_DIM * K_DIM;
  float*  ca  = (float*)(Wbb + (size_t)H_DIM * K_DIM);
  float*  cb  = ca + (size_t)B_DIM * H_DIM * NCHUNK;
  float*  carry = cb + (size_t)B_DIM * H_DIM * NCHUNK;

  // 0) bulk f32 -> bf16 conversions (8 elems/thread)
  f32_to_bf16_kernel<<<(M_DIM * (size_t)K_DIM / 8) / 256, 256, 0, stream>>>(x,  (unsigned int*)xbp);
  f32_to_bf16_kernel<<<((size_t)H_DIM * K_DIM / 8) / 256, 256, 0, stream>>>(Wd, (unsigned int*)Wdb);
  f32_to_bf16_kernel<<<((size_t)H_DIM * K_DIM / 8) / 256, 256, 0, stream>>>(Wb, (unsigned int*)Wbb);

  // 1) dual GEMM + epilogue: At -> ws, bt -> d_out
  dim3 gg(H_DIM / 128, M_DIM / 128);
  liquid_proj_wmma<<<gg, 256, 0, stream>>>(xbp, Wdb, Wbb, bd, bb, A_log, At, out);

  // 2) per-chunk affine composition
  chunk_reduce<<<(B_DIM * H_DIM * NCHUNK) / 256, 256, 0, stream>>>(At, out, ca, cb);

  // 3) inter-chunk scan (uses h0)
  chunk_scan<<<(B_DIM * H_DIM) / 256, 256, 0, stream>>>(ca, cb, h0, carry);

  // 4) apply within chunks, tanh, overwrite d_out
  chunk_apply<<<(B_DIM * H_DIM * NCHUNK) / 256, 256, 0, stream>>>(At, carry, out);
}
